// MultiHeadLatentAttention_57939108823111
// MI455X (gfx1250) — matrix operand order repaired and checked by reading, not compiled, not run on hardware
//
#include <hip/hip_runtime.h>
#include <hip/hip_bf16.h>

// ---------------------------------------------------------------------------
// MLA forward for MI455X (gfx1250, wave32): f16 WMMA everywhere, TDM-staged
// double-buffered GEMM *and* flash attention (Tensor Data Mover + TENSORcnt).
// ---------------------------------------------------------------------------

typedef __attribute__((ext_vector_type(16))) _Float16     v16h;
typedef __attribute__((ext_vector_type(8)))  float        v8f;
typedef __attribute__((ext_vector_type(4)))  unsigned int u32x4;
typedef __attribute__((ext_vector_type(8)))  unsigned int u32x8;

__device__ __forceinline__ v8f zero8() {
    v8f z;
#pragma unroll
    for (int i = 0; i < 8; ++i) z[i] = 0.0f;
    return z;
}

__device__ __forceinline__ v16h frag2(const _Float16* p0, const _Float16* p1) {
    union { v16h v; uint4 u[2]; } f;
    f.u[0] = *(const uint4*)p0;
    f.u[1] = *(const uint4*)p1;
    return f.v;
}

__device__ __forceinline__ v8f wmma_f16(v16h a, v16h b, v8f c) {
    return __builtin_amdgcn_wmma_f32_16x16x32_f16(false, a, false, b,
                                                  (short)0, c, false, false);
}

// Low 32 bits of a flat shared-aperture address = LDS byte offset.
__device__ __forceinline__ unsigned lds_byte_off(const void* p) {
    return (unsigned)(size_t)p;
}

// Pad configs for D# group1 word0 (pad_enable<<20 | interval<<22 | amount<<25)
//  - GEMM tiles : 32-half rows -> 48-half stride (pad 32B after each 64B row)
//  - V^T tiles  : 64-half rows -> 72-half stride (pad 16B after each 128B row)
#define PAD_NONE  0u
#define PAD_GEMM  ((1u << 20) | (3u << 22) | (7u << 25))
#define PAD_VT    ((1u << 20) | (4u << 22) | (3u << 25))

// ---------------------------------------------------------------------------
// Tensor Data Mover: 2D f16 tile load Global->LDS (ISA 08_async_tensor D#).
// One instruction per wave; EXEC ignored; tracked by TENSORcnt.
// ---------------------------------------------------------------------------
__device__ __forceinline__ void tdm_load_tile_f16(
    unsigned lds_off, const _Float16* gptr,
    unsigned tensor_d0, unsigned tensor_d1,      // tensor extent (elems)
    unsigned tile_d0, unsigned tile_d1,          // tile extent (elems)
    unsigned stride_d0,                          // row stride (elems)
    unsigned pad_bits)
{
    unsigned long long ga = (unsigned long long)(size_t)gptr;
    u32x4 g0;
    g0[0] = 1u;                                   // count=1, user mode
    g0[1] = lds_off;                              // lds_addr (bytes)
    g0[2] = (unsigned)(ga & 0xffffffffu);         // global_addr[31:0]
    g0[3] = (unsigned)((ga >> 32) & 0x01ffffffu)  // global_addr[56:32]
          | (2u << 30);                           // type = 2 (image)
    u32x8 g1;
    g1[0] = (1u << 16) | pad_bits;                // data_size = 2 bytes
    g1[1] = (tensor_d0 & 0xffffu) << 16;          // tensor_dim0[15:0]
    g1[2] = (tensor_d0 >> 16) | ((tensor_d1 & 0xffffu) << 16);
    g1[3] = (tensor_d1 >> 16) | (tile_d0 << 16);  // tile_dim0
    g1[4] = tile_d1 & 0xffffu;                    // tile_dim1 (tile_dim2=0)
    g1[5] = stride_d0;                            // tensor_dim0_stride[31:0]
    g1[6] = 0u;
    g1[7] = 0u;
    asm volatile("tensor_load_to_lds %0, %1" :: "s"(g0), "s"(g1) : "memory");
}

// ---------------------------------------------------------------------------
// fp32 -> f16 bulk convert (row-major copy).
// ---------------------------------------------------------------------------
__global__ __launch_bounds__(256) void cvt_f16(const float* __restrict__ src,
                                               _Float16* __restrict__ dst, int n4)
{
    int idx = blockIdx.x * 256 + threadIdx.x;
    if (idx >= n4) return;
    float4 f = *(const float4*)&src[(size_t)idx * 4];
    union { _Float16 h[4]; uint2 u; } cv;
    cv.h[0] = (_Float16)f.x; cv.h[1] = (_Float16)f.y;
    cv.h[2] = (_Float16)f.z; cv.h[3] = (_Float16)f.w;
    *(uint2*)&dst[(size_t)idx * 4] = cv.u;
}

// ---------------------------------------------------------------------------
// W (K x N, fp32) -> Wt (N x K, f16). 32x32 tiles via LDS.
// ---------------------------------------------------------------------------
__global__ __launch_bounds__(256) void transpose_cvt(const float* __restrict__ W,
    _Float16* __restrict__ Wt, int K, int N)
{
    __shared__ _Float16 t[32][36];
    const int k0 = blockIdx.x * 32, n0 = blockIdx.y * 32;
    const int tid = threadIdx.x;
    const int r = tid >> 3, c4 = (tid & 7) * 4;
    float4 f = *(const float4*)&W[(size_t)(k0 + r) * N + n0 + c4];
    t[r][c4 + 0] = (_Float16)f.x; t[r][c4 + 1] = (_Float16)f.y;
    t[r][c4 + 2] = (_Float16)f.z; t[r][c4 + 3] = (_Float16)f.w;
    __syncthreads();
    union { _Float16 h[4]; uint2 u; } cv;
    cv.h[0] = t[c4 + 0][r]; cv.h[1] = t[c4 + 1][r];
    cv.h[2] = t[c4 + 2][r]; cv.h[3] = t[c4 + 3][r];
    *(uint2*)&Wt[(size_t)(n0 + r) * K + k0 + c4] = cv.u;
}

// ---------------------------------------------------------------------------
// v (L x 2048, f16) -> vt (16 x 128 x L, f16): per-head transpose.
// ---------------------------------------------------------------------------
__global__ __launch_bounds__(256) void transpose_v(const _Float16* __restrict__ v,
    _Float16* __restrict__ vt, int L)
{
    __shared__ _Float16 t[32][36];
    const int l0 = blockIdx.x * 32, d0 = blockIdx.y * 32, h = blockIdx.z;
    const int tid = threadIdx.x;
    const int r = tid >> 3, c4 = (tid & 7) * 4;
    union { _Float16 h4[4]; uint2 u; } cv;
    cv.u = *(const uint2*)&v[(size_t)(l0 + r) * 2048 + h * 128 + d0 + c4];
    t[r][c4 + 0] = cv.h4[0]; t[r][c4 + 1] = cv.h4[1];
    t[r][c4 + 2] = cv.h4[2]; t[r][c4 + 3] = cv.h4[3];
    __syncthreads();
    cv.h4[0] = t[c4 + 0][r]; cv.h4[1] = t[c4 + 1][r];
    cv.h4[2] = t[c4 + 2][r]; cv.h4[3] = t[c4 + 3][r];
    *(uint2*)&vt[((size_t)h * 128 + d0 + r) * L + l0 + c4] = cv.u;
}

// ---------------------------------------------------------------------------
// GEMM: C[M,N] = A[M,K] * Bt[N,K]^T, f16 operands, f32 accumulate.
// 128x128x32 tiles; 8 waves (4x2), each 32x64 (2x4 WMMA tiles).
// TDM-staged, double-buffered LDS (wave 0 drives the DMA engine).
// ---------------------------------------------------------------------------
#define BM 128
#define BN 128
#define BK 32
#define LDT 48   // padded f16 stride produced by the TDM pad fields

template <bool OUT_F16>
__global__ __launch_bounds__(256) void gemm_f16_wmma(
    const _Float16* __restrict__ A, const _Float16* __restrict__ Bt,
    void* __restrict__ Cout, int M, int N, int K)
{
    __shared__ _Float16 a_lds[2][BM * LDT];
    __shared__ _Float16 b_lds[2][BN * LDT];

    const int tid  = threadIdx.x;
    const int lane = tid & 31;
    const int wave = tid >> 5;
    const int wm   = wave >> 1;
    const int wn   = wave & 1;
    const int bm   = blockIdx.x * BM;
    const int bn   = blockIdx.y * BN;

    const int lrow  = lane & 15;
    const int kba   = (lane & 16) ? 8  : 0;
    const int kbb   = (lane & 16) ? 8  : 0;
    const int rbase = (lane >> 4) * 8;

    v8f acc[2][4];
#pragma unroll
    for (int i = 0; i < 2; ++i)
#pragma unroll
        for (int j = 0; j < 4; ++j) acc[i][j] = zero8();

    const int nk = K / BK;
    if (wave == 0) {
        tdm_load_tile_f16(lds_byte_off(&a_lds[0][0]), A + (size_t)bm * K,
                          (unsigned)K, (unsigned)M, BK, BM, (unsigned)K, PAD_GEMM);
        tdm_load_tile_f16(lds_byte_off(&b_lds[0][0]), Bt + (size_t)bn * K,
                          (unsigned)K, (unsigned)N, BK, BN, (unsigned)K, PAD_GEMM);
    }

    for (int i = 0; i < nk; ++i) {
        if (wave == 0) __builtin_amdgcn_s_wait_tensorcnt(0);
        __syncthreads();
        if (wave == 0 && i + 1 < nk) {
            int k1 = (i + 1) * BK;
            int nb = (i + 1) & 1;
            tdm_load_tile_f16(lds_byte_off(&a_lds[nb][0]),
                              A + (size_t)bm * K + k1,
                              (unsigned)K, (unsigned)M, BK, BM, (unsigned)K, PAD_GEMM);
            tdm_load_tile_f16(lds_byte_off(&b_lds[nb][0]),
                              Bt + (size_t)bn * K + k1,
                              (unsigned)K, (unsigned)N, BK, BN, (unsigned)K, PAD_GEMM);
        }
        const _Float16* al = a_lds[i & 1];
        const _Float16* bl = b_lds[i & 1];

        v16h af[2], bf[4];
#pragma unroll
        for (int ii = 0; ii < 2; ++ii) {
            const _Float16* p = &al[(wm * 32 + ii * 16 + lrow) * LDT + kba];
            af[ii] = frag2(p, p + 16);
        }
#pragma unroll
        for (int j = 0; j < 4; ++j) {
            const _Float16* p = &bl[(wn * 64 + j * 16 + lrow) * LDT + kbb];
            bf[j] = frag2(p, p + 16);
        }
#pragma unroll
        for (int ii = 0; ii < 2; ++ii)
#pragma unroll
            for (int j = 0; j < 4; ++j)
                acc[ii][j] = wmma_f16(af[ii], bf[j], acc[ii][j]);
    }

    // epilogue (ISA C/D layout: lanes 0-15 M=vgpr, lanes 16-31 M=vgpr+8)
#pragma unroll
    for (int i = 0; i < 2; ++i)
#pragma unroll
        for (int j = 0; j < 4; ++j)
#pragma unroll
            for (int r = 0; r < 8; ++r) {
                int row = bm + wm * 32 + i * 16 + rbase + r;
                int col = bn + wn * 64 + j * 16 + (lane & 15);
                if (OUT_F16)
                    ((_Float16*)Cout)[(size_t)row * N + col] = (_Float16)acc[i][j][r];
                else
                    ((float*)Cout)[(size_t)row * N + col] = acc[i][j][r];
            }
}

// ---------------------------------------------------------------------------
// RoPE in-place on f16 X of shape (L, 16*192); last 64 dims of each head.
// ---------------------------------------------------------------------------
__global__ __launch_bounds__(256) void rope_f16(_Float16* __restrict__ X, int total)
{
    int idx = blockIdx.x * 256 + threadIdx.x;
    if (idx >= total) return;
    int i = idx & 31;
    int h = (idx >> 5) & 15;
    int l = idx >> 9;
    float inv_freq = __expf(-(float)i * (9.210340371976184f / 32.0f)); // ln(1e4)/32
    float ang = (float)l * inv_freq;
    float c = __cosf(ang), s = __sinf(ang);
    _Float16* base = &X[(size_t)l * 3072 + h * 192 + 128];
    float a = (float)base[i], b = (float)base[i + 32];
    base[i]      = (_Float16)(a * c - b * s);
    base[i + 32] = (_Float16)(b * c + a * s);
}

// ---------------------------------------------------------------------------
// Flash attention (causal): one head x 64-query block per workgroup.
// Q,K: (L, 16*192) f16 (RoPE applied); Vt: (16, 128, L) f16.
// TDM-staged, double-buffered K and V^T tiles selected by *offset arithmetic*
// from one LDS base (keeps address-space inference -> ds_load, not flat).
// ---------------------------------------------------------------------------
#define FA_QSZ (64 * 192)    // Q tile, halfs
#define FA_KSZ (64 * 192)    // K tile, halfs
#define FA_VSZ (128 * 72)    // V^T tile (padded stride 72), halfs
#define FA_PSZ (64 * 80)     // P tile (stride 80), halfs

__global__ __launch_bounds__(256) void mla_flash_attn(
    const _Float16* __restrict__ Q, const _Float16* __restrict__ Kmat,
    const _Float16* __restrict__ Vt, _Float16* __restrict__ O, int L)
{
    extern __shared__ char smem_raw[];
    _Float16* lds_h  = (_Float16*)smem_raw;
    _Float16* q_lds  = lds_h;                                  // FA_QSZ
    _Float16* k_base = lds_h + FA_QSZ;                         // 2 * FA_KSZ
    _Float16* v_base = k_base + 2 * FA_KSZ;                    // 2 * FA_VSZ
    _Float16* p_lds  = v_base + 2 * FA_VSZ;                    // FA_PSZ
    float*    s_lds  = (float*)(p_lds + FA_PSZ);               // 64 x 64
    float*    m_state = s_lds + 64 * 64;
    float*    l_state = m_state + 64;
    float*    corr_s  = l_state + 64;

    const int h  = blockIdx.x;
    const int qb = blockIdx.y;
    const int q0 = qb * 64;
    const int tid  = threadIdx.x;
    const int lane = tid & 31;
    const int wave = tid >> 5;
    const int wm   = wave >> 1;
    const int wn   = wave & 1;
    const int lrow  = lane & 15;
    const int kba   = (lane & 16) ? 8  : 0;
    const int kbb   = (lane & 16) ? 8  : 0;
    const int rbase = (lane >> 4) * 8;
    const float scale = 0.07216878364870323f;  // 1/sqrt(192)

    v8f oacc[4];
#pragma unroll
    for (int j = 0; j < 4; ++j) oacc[j] = zero8();

    if (tid < 64) { m_state[tid] = -1.0e30f; l_state[tid] = 0.0f; }

    const unsigned k_lds_off = lds_byte_off(k_base);
    const unsigned v_lds_off = lds_byte_off(v_base);

    if (wave == 0) {
        tdm_load_tile_f16(lds_byte_off(q_lds), Q + (size_t)q0 * 3072 + h * 192,
                          3072u, (unsigned)L, 192u, 64u, 3072u, PAD_NONE);
        tdm_load_tile_f16(k_lds_off, Kmat + (size_t)h * 192,
                          3072u, (unsigned)L, 192u, 64u, 3072u, PAD_NONE);
        tdm_load_tile_f16(v_lds_off, Vt + (size_t)h * 128 * L,
                          (unsigned)L, 128u, 64u, 128u, (unsigned)L, PAD_VT);
    }

    for (int kb = 0; kb <= qb; ++kb) {
        if (wave == 0) __builtin_amdgcn_s_wait_tensorcnt(0);
        __syncthreads();
        if (wave == 0 && kb + 1 <= qb) {
            unsigned nb = (unsigned)(kb + 1) & 1u;
            tdm_load_tile_f16(k_lds_off + nb * (FA_KSZ * 2),
                              Kmat + (size_t)(kb + 1) * 64 * 3072 + h * 192,
                              3072u, (unsigned)L, 192u, 64u, 3072u, PAD_NONE);
            tdm_load_tile_f16(v_lds_off + nb * (FA_VSZ * 2),
                              Vt + (size_t)h * 128 * L + (kb + 1) * 64,
                              (unsigned)L, 128u, 64u, 128u, (unsigned)L, PAD_VT);
        }
        const _Float16* kl = k_base + (kb & 1) * FA_KSZ;
        const _Float16* vl = v_base + (kb & 1) * FA_VSZ;

        // S = Q K^T: wave -> rows wm*16..+15, cols wn*32..+31
        v8f sacc[2];
        sacc[0] = zero8(); sacc[1] = zero8();
#pragma unroll
        for (int ks = 0; ks < 6; ++ks) {
            const _Float16* ap = &q_lds[(wm * 16 + lrow) * 192 + ks * 32 + kba];
            v16h af = frag2(ap, ap + 16);
#pragma unroll
            for (int j = 0; j < 2; ++j) {
                const _Float16* bp = &kl[(wn * 32 + j * 16 + lrow) * 192 + ks * 32 + kbb];
                v16h bf = frag2(bp, bp + 16);
                sacc[j] = wmma_f16(af, bf, sacc[j]);
            }
        }
#pragma unroll
        for (int j = 0; j < 2; ++j)
#pragma unroll
            for (int r = 0; r < 8; ++r)
                s_lds[(wm * 16 + rbase + r) * 64 + wn * 32 + j * 16 + (lane & 15)] = sacc[j][r];
        __syncthreads();

        // online softmax: 4 threads per query row
        {
            int row   = tid >> 2;
            int cbase = (tid & 3) * 16;
            int qrow  = q0 + row;
            int kg0   = kb * 64 + cbase;
            const float* srow = &s_lds[row * 64 + cbase];
            float mx = -1.0e30f;
            float sv[16];
#pragma unroll
            for (int c = 0; c < 16; ++c) {
                float s = (kg0 + c > qrow) ? -1.0e9f : srow[c] * scale;
                sv[c] = s;
                mx = fmaxf(mx, s);
            }
            mx = fmaxf(mx, __shfl_xor(mx, 1, 4));
            mx = fmaxf(mx, __shfl_xor(mx, 2, 4));
            float mold = m_state[row];
            float mnew = fmaxf(mold, mx);
            float sum = 0.0f;
#pragma unroll
            for (int c = 0; c < 16; ++c) {
                float p = __expf(sv[c] - mnew);
                sum += p;
                p_lds[row * 80 + cbase + c] = (_Float16)p;
            }
            sum += __shfl_xor(sum, 1, 4);
            sum += __shfl_xor(sum, 2, 4);
            if ((tid & 3) == 0) {
                float corr = __expf(mold - mnew);
                corr_s[row]  = corr;
                m_state[row] = mnew;
                l_state[row] = l_state[row] * corr + sum;
            }
        }
        __syncthreads();

        // O = O*corr + P V : wave -> rows wm*16..+15, cols wn*64..+63
        float cf[8];
#pragma unroll
        for (int r = 0; r < 8; ++r) cf[r] = corr_s[wm * 16 + rbase + r];
#pragma unroll
        for (int j = 0; j < 4; ++j)
#pragma unroll
            for (int r = 0; r < 8; ++r) oacc[j][r] *= cf[r];
#pragma unroll
        for (int ks = 0; ks < 2; ++ks) {
            const _Float16* ap = &p_lds[(wm * 16 + lrow) * 80 + ks * 32 + kba];
            v16h af = frag2(ap, ap + 16);
#pragma unroll
            for (int j = 0; j < 4; ++j) {
                const _Float16* bp = &vl[(wn * 64 + j * 16 + lrow) * 72 + ks * 32 + kbb];
                v16h bf = frag2(bp, bp + 16);
                oacc[j] = wmma_f16(af, bf, oacc[j]);
            }
        }
        __syncthreads();   // all waves done with this buffer before next DMA
    }

    // normalize and write f16 (L, 16*128)
#pragma unroll
    for (int j = 0; j < 4; ++j)
#pragma unroll
        for (int r = 0; r < 8; ++r) {
            int row = wm * 16 + rbase + r;
            int col = wn * 64 + j * 16 + (lane & 15);
            float inv = 1.0f / l_state[row];
            O[(size_t)(q0 + row) * 2048 + h * 128 + col] = (_Float16)(oacc[j][r] * inv);
        }
}

// ---------------------------------------------------------------------------
// Host-side orchestration
// ---------------------------------------------------------------------------
extern "C" void kernel_launch(void* const* d_in, const int* in_sizes, int n_in,
                              void* d_out, int out_size, void* d_ws, size_t ws_size,
                              hipStream_t stream)
{
    (void)in_sizes; (void)n_in; (void)out_size; (void)ws_size;
    const float* hidden   = (const float*)d_in[0];
    /* d_in[1] = attention_mask: pure causal (-1e9) -> folded into attention */
    const float* Wq_down  = (const float*)d_in[2];   // 2048 x 1536
    const float* Wq_up    = (const float*)d_in[3];   // 1536 x 3072
    const float* Wkv_down = (const float*)d_in[4];   // 2048 x 512
    const float* Wk_up    = (const float*)d_in[5];   // 512  x 3072
    const float* Wv_up    = (const float*)d_in[6];   // 512  x 2048
    const float* Wo       = (const float*)d_in[7];   // 2048 x 2048
    float* out = (float*)d_out;

    const int L = 2048, D = 2048, DQ = 1536, DC = 512, HD = 3072, HV = 2048;

    _Float16* ws = (_Float16*)d_ws;
    _Float16* h16    = ws;                               // L x D
    _Float16* wqd_t  = h16   + (size_t)L * D;            // DQ x D
    _Float16* wqu_t  = wqd_t + (size_t)DQ * D;           // HD x DQ
    _Float16* wkvd_t = wqu_t + (size_t)HD * DQ;          // DC x D
    _Float16* wku_t  = wkvd_t + (size_t)DC * D;          // HD x DC
    _Float16* wvu_t  = wku_t + (size_t)HD * DC;          // HV x DC
    _Float16* wo_t   = wvu_t + (size_t)HV * DC;          // D x D
    _Float16* tq     = wo_t  + (size_t)D * D;            // L x DQ
    _Float16* q      = tq    + (size_t)L * DQ;           // L x HD
    _Float16* ckv    = q     + (size_t)L * HD;           // L x DC
    _Float16* k      = ckv   + (size_t)L * DC;           // L x HD
    _Float16* v      = k     + (size_t)L * HD;           // L x HV
    _Float16* vt     = v     + (size_t)L * HV;           // 16 x 128 x L
    _Float16* attn   = vt    + (size_t)L * HV;           // L x HV

    dim3 blk(256);

    // one-time operand preparation (f16, weights transposed)
    cvt_f16<<<(L * D / 4 + 255) / 256, blk, 0, stream>>>(hidden, h16, L * D / 4);
    transpose_cvt<<<dim3(D / 32, DQ / 32), blk, 0, stream>>>(Wq_down, wqd_t, D, DQ);
    transpose_cvt<<<dim3(DQ / 32, HD / 32), blk, 0, stream>>>(Wq_up, wqu_t, DQ, HD);
    transpose_cvt<<<dim3(D / 32, DC / 32), blk, 0, stream>>>(Wkv_down, wkvd_t, D, DC);
    transpose_cvt<<<dim3(DC / 32, HD / 32), blk, 0, stream>>>(Wk_up, wku_t, DC, HD);
    transpose_cvt<<<dim3(DC / 32, HV / 32), blk, 0, stream>>>(Wv_up, wvu_t, DC, HV);
    transpose_cvt<<<dim3(D / 32, D / 32), blk, 0, stream>>>(Wo, wo_t, D, D);

    // projections (all f16-out, TDM-staged WMMA GEMMs)
    gemm_f16_wmma<true><<<dim3(L / BM, DQ / BN), blk, 0, stream>>>(h16, wqd_t, tq, L, DQ, D);
    gemm_f16_wmma<true><<<dim3(L / BM, HD / BN), blk, 0, stream>>>(tq, wqu_t, q, L, HD, DQ);
    gemm_f16_wmma<true><<<dim3(L / BM, DC / BN), blk, 0, stream>>>(h16, wkvd_t, ckv, L, DC, D);
    gemm_f16_wmma<true><<<dim3(L / BM, HD / BN), blk, 0, stream>>>(ckv, wku_t, k, L, HD, DC);
    gemm_f16_wmma<true><<<dim3(L / BM, HV / BN), blk, 0, stream>>>(ckv, wvu_t, v, L, HV, DC);

    int nrope = L * 16 * 32;
    rope_f16<<<(nrope + 255) / 256, blk, 0, stream>>>(q, nrope);
    rope_f16<<<(nrope + 255) / 256, blk, 0, stream>>>(k, nrope);

    transpose_v<<<dim3(L / 32, 4, 16), blk, 0, stream>>>(v, vt, L);

    size_t attn_lds = (size_t)(FA_QSZ + 2 * FA_KSZ + 2 * FA_VSZ + FA_PSZ) * sizeof(_Float16)
                    + (size_t)(64 * 64 + 3 * 64) * sizeof(float);
    mla_flash_attn<<<dim3(16, L / 64), blk, attn_lds, stream>>>(q, k, vt, attn, L);

    // output projection in fp32
    gemm_f16_wmma<false><<<dim3(L / BM, D / BN), blk, 0, stream>>>(attn, wo_t, out, L, D, D);
}
